// BlockDiffusionDecoder_9732395892789
// MI455X (gfx1250) — matrix operand order repaired and checked by reading, not compiled, not run on hardware
//
#include <hip/hip_runtime.h>

// ---------------- problem constants (match reference) ----------------
constexpr int Bc = 4, Lc = 512, Tc = 1024, Dc = 1024, Hc = 16, HDc = 64;
constexpr int NLc = 2, DFFc = 4096, Vc = 32001, Sc = 128;

typedef __attribute__((ext_vector_type(16))) __bf16 v16bf;
typedef __attribute__((ext_vector_type(8)))  float v8f;
typedef __attribute__((ext_vector_type(4)))  int   v4i;

struct __align__(16) us8 { unsigned short v[8]; };

#define DEV __device__ __forceinline__

#ifndef __has_builtin
#define __has_builtin(x) 0
#endif

#if __has_builtin(__builtin_amdgcn_global_load_async_to_lds_b128)
#define HAVE_ASYNC 1
#else
#define HAVE_ASYNC 0
#endif

#if __has_builtin(__builtin_amdgcn_s_wait_asynccnt)
#define WAIT_ASYNC0() __builtin_amdgcn_s_wait_asynccnt(0)
#else
#define WAIT_ASYNC0() asm volatile("s_wait_asynccnt 0" ::: "memory")
#endif

DEV float bf2f(unsigned short s) {
    union { unsigned int u; float f; } c; c.u = ((unsigned int)s) << 16; return c.f;
}
DEV unsigned short f2bf(float f) {
    union { float f; unsigned int u; } c; c.f = f;
    unsigned int u = c.u;
    unsigned int r = (u + 0x7FFFu + ((u >> 16) & 1u)) >> 16;   // RNE
    return (unsigned short)r;
}
DEV __bf16 us2bf(unsigned short s) {
    union { unsigned short u; __bf16 b; } c; c.u = s; return c.b;
}

// stage one 16B chunk: global -> LDS (async if available)
DEV void cp16(unsigned short* ldst, const unsigned short* gsrc) {
#if HAVE_ASYNC
    __builtin_amdgcn_global_load_async_to_lds_b128(
        (__attribute__((address_space(1))) v4i*)gsrc,
        (__attribute__((address_space(3))) v4i*)ldst, 0, 0);
#else
    *(us8*)ldst = *(const us8*)gsrc;
#endif
}

// ---------------- elementwise helpers ----------------
__global__ void cvt_f32_bf16_kernel(const float* __restrict__ src,
                                    unsigned short* __restrict__ dst, size_t n) {
    size_t i = (size_t)blockIdx.x * 256 + threadIdx.x;
    if (i < n) dst[i] = f2bf(src[i]);
}

// embedding gather + scale + pos add + LayerNorm -> fp32 residual stream h
__global__ void embed_ln_kernel(const int* __restrict__ ids,
                                const float* __restrict__ emb,
                                const float* __restrict__ pos,
                                const float* __restrict__ gam,
                                const float* __restrict__ bet,
                                float* __restrict__ out) {
    int row = blockIdx.x;                 // 0..B*T-1
    int t   = row % Tc;
    int id  = ids[row];
    const float* e = emb + (size_t)id * Dc;
    const float* p = pos + (size_t)t  * Dc;
    float v[4];
    float s = 0.f;
#pragma unroll
    for (int j = 0; j < 4; ++j) {
        int d = threadIdx.x + j * 256;
        v[j] = e[d] * 32.0f + p[d];       // sqrt(D)=32
        s += v[j];
    }
    __shared__ float red[256];
    red[threadIdx.x] = s; __syncthreads();
    for (int st = 128; st > 0; st >>= 1) {
        if (threadIdx.x < st) red[threadIdx.x] += red[threadIdx.x + st];
        __syncthreads();
    }
    float mu = red[0] * (1.0f / Dc);
    __syncthreads();
    float ss = 0.f;
#pragma unroll
    for (int j = 0; j < 4; ++j) { float d = v[j] - mu; ss += d * d; }
    red[threadIdx.x] = ss; __syncthreads();
    for (int st = 128; st > 0; st >>= 1) {
        if (threadIdx.x < st) red[threadIdx.x] += red[threadIdx.x + st];
        __syncthreads();
    }
    float inv = rsqrtf(red[0] * (1.0f / Dc) + 1e-5f);
#pragma unroll
    for (int j = 0; j < 4; ++j) {
        int d = threadIdx.x + j * 256;
        out[(size_t)row * Dc + d] = (v[j] - mu) * inv * gam[d] + bet[d];
    }
}

// row LayerNorm over D=1024, fp32 in -> bf16 out (GEMM A operand)
__global__ void layernorm_bf16_kernel(const float* __restrict__ x,
                                      const float* __restrict__ gam,
                                      const float* __restrict__ bet,
                                      unsigned short* __restrict__ out) {
    int row = blockIdx.x;
    const float* xr = x + (size_t)row * Dc;
    float v[4];
    float s = 0.f;
#pragma unroll
    for (int j = 0; j < 4; ++j) { v[j] = xr[threadIdx.x + j * 256]; s += v[j]; }
    __shared__ float red[256];
    red[threadIdx.x] = s; __syncthreads();
    for (int st = 128; st > 0; st >>= 1) {
        if (threadIdx.x < st) red[threadIdx.x] += red[threadIdx.x + st];
        __syncthreads();
    }
    float mu = red[0] * (1.0f / Dc);
    __syncthreads();
    float ss = 0.f;
#pragma unroll
    for (int j = 0; j < 4; ++j) { float d = v[j] - mu; ss += d * d; }
    red[threadIdx.x] = ss; __syncthreads();
    for (int st = 128; st > 0; st >>= 1) {
        if (threadIdx.x < st) red[threadIdx.x] += red[threadIdx.x + st];
        __syncthreads();
    }
    float inv = rsqrtf(red[0] * (1.0f / Dc) + 1e-5f);
#pragma unroll
    for (int j = 0; j < 4; ++j) {
        int d = threadIdx.x + j * 256;
        out[(size_t)row * Dc + d] = f2bf((v[j] - mu) * inv * gam[d] + bet[d]);
    }
}

// (B,Tq,D) f32 -> (B,H,Tq,HD) bf16, optional scale (1/sqrt(HD) for q)
__global__ void split_heads_kernel(const float* __restrict__ src,
                                   unsigned short* __restrict__ dst,
                                   int Tq, float scale) {
    size_t i = (size_t)blockIdx.x * 256 + threadIdx.x;
    size_t total = (size_t)Bc * Tq * Dc;
    if (i >= total) return;
    int d = (int)(i % Dc);
    size_t bt = i / Dc;
    int t = (int)(bt % Tq);
    int b = (int)(bt / Tq);
    int h = d / HDc, hd = d % HDc;
    dst[(((size_t)b * Hc + h) * Tq + t) * HDc + hd] = f2bf(src[i] * scale);
}

// (B,Tq,D) f32 -> (B,H,HD,Tq) bf16  (V transposed so AV GEMM sees (N=HD,K=Tq))
__global__ void split_heads_T_kernel(const float* __restrict__ src,
                                     unsigned short* __restrict__ dst, int Tq) {
    size_t i = (size_t)blockIdx.x * 256 + threadIdx.x;
    size_t total = (size_t)Bc * Tq * Dc;
    if (i >= total) return;
    int d = (int)(i % Dc);
    size_t bt = i / Dc;
    int t = (int)(bt % Tq);
    int b = (int)(bt / Tq);
    int h = d / HDc, hd = d % HDc;
    dst[(((size_t)b * Hc + h) * HDc + hd) * Tq + t] = f2bf(src[i]);
}

// (B,H,T,HD) f32 -> (B,T,D) bf16
__global__ void merge_heads_kernel(const float* __restrict__ src,
                                   unsigned short* __restrict__ dst) {
    size_t i = (size_t)blockIdx.x * 256 + threadIdx.x;
    size_t total = (size_t)Bc * Tc * Dc;
    if (i >= total) return;
    int d = (int)(i % Dc);
    size_t bt = i / Dc;
    int t = (int)(bt % Tc);
    int b = (int)(bt / Tc);
    int h = d / HDc, hd = d % HDc;
    dst[i] = f2bf(src[(((size_t)b * Hc + h) * Tc + t) * HDc + hd]);
}

// in-place row softmax on bf16 buffer, rows of length N
__global__ void softmax_bf16_kernel(unsigned short* __restrict__ p, int N) {
    size_t row = blockIdx.x;
    unsigned short* pr = p + row * (size_t)N;
    __shared__ float red[256];
    float mx = -3.0e38f;
    for (int j = threadIdx.x; j < N; j += 256) mx = fmaxf(mx, bf2f(pr[j]));
    red[threadIdx.x] = mx; __syncthreads();
    for (int st = 128; st > 0; st >>= 1) {
        if (threadIdx.x < st) red[threadIdx.x] = fmaxf(red[threadIdx.x], red[threadIdx.x + st]);
        __syncthreads();
    }
    mx = red[0]; __syncthreads();
    float sum = 0.f;
    for (int j = threadIdx.x; j < N; j += 256) {
        float e = expf(bf2f(pr[j]) - mx);
        pr[j] = f2bf(e);
        sum += e;
    }
    red[threadIdx.x] = sum; __syncthreads();
    for (int st = 128; st > 0; st >>= 1) {
        if (threadIdx.x < st) red[threadIdx.x] += red[threadIdx.x + st];
        __syncthreads();
    }
    float inv = 1.0f / red[0];
    for (int j = threadIdx.x; j < N; j += 256) pr[j] = f2bf(bf2f(pr[j]) * inv);
}

// ---------------- batched bf16 WMMA GEMM: C = A(M,K) x B(N,K)^T ----------------
// Workgroup: 256 threads / 8 waves, C-tile 128(M) x 64(N).
// Per k-step(32): A-tile 128x32 + B-tile 64x32 staged to double-buffered LDS via
// async global->LDS copies (ASYNCcnt); each wave computes a 16x64 strip: all
// fragments are fetched from LDS first (one DScnt drain), then 4 independent
// WMMAs issue back-to-back reusing one A fragment.
// Requires: M % 128 == 0, K % 32 == 0 (true for every call site); N guarded.
// EPI 0: f32 out, +bias (if non-null), +addend (residual, if non-null)
// EPI 1: bf16 out, bias then tanh-GELU
// EPI 2: bf16 out, + BD3LM self-attention mask (computed from indices)
// EPI 3: bf16 out, + cross-attention mask from enc_mask (batch = z/H)
template <int EPI>
__global__ void __launch_bounds__(256)
wmma_gemm_kernel(const unsigned short* __restrict__ A,
                 const unsigned short* __restrict__ Bw,
                 const float* __restrict__ bias,
                 const float* __restrict__ addend,
                 float* __restrict__ Cf,
                 unsigned short* __restrict__ Cb,
                 const int* __restrict__ emask,
                 int M, int N, int K,
                 long long strA, long long strB, long long strC) {
    __shared__ unsigned short lA[2][128 * 32];   // 8 KB per buffer
    __shared__ unsigned short lB[2][64 * 32];    // 4 KB per buffer

    const int tid  = threadIdx.x;
    const int lane = tid & 31;
    const int wave = tid >> 5;                   // 0..7, one 16-row strip each
    const int hi   = lane >> 4;
    const int ln15 = lane & 15;
    const int tileM = blockIdx.y * 128;
    const int tileN = blockIdx.x * 64;
    const int z = blockIdx.z;

    const unsigned short* Ag = A  + (size_t)z * strA;
    const unsigned short* Bg = Bw + (size_t)z * strB;

    // staging map: thread -> (row 0..63, 16B chunk) ; 3 chunks per thread
    const int rowS = tid >> 2;                   // 0..63
    const int koS  = (tid & 3) * 8;              // element offset 0/8/16/24
    int nClamp = tileN + rowS; if (nClamp > N - 1) nClamp = N - 1;

    auto stage = [&](int k0, int buf) {
        cp16(&lA[buf][rowS * 32 + koS],
             Ag + (size_t)(tileM + rowS) * K + k0 + koS);
        cp16(&lA[buf][(64 + rowS) * 32 + koS],
             Ag + (size_t)(tileM + 64 + rowS) * K + k0 + koS);
        cp16(&lB[buf][rowS * 32 + koS],
             Bg + (size_t)nClamp * K + k0 + koS);
    };

    v8f acc[4] = {};
    const int nk = K >> 5;
    stage(0, 0);
    int buf = 0;
    for (int ks = 0; ks < nk; ++ks) {
#if HAVE_ASYNC
        WAIT_ASYNC0();
#endif
        __syncthreads();
        if (ks + 1 < nk) stage((ks + 1) * 32, buf ^ 1);

        // ---- fetch ALL fragments from LDS first (single DScnt drain) ----
        // A fragment for this wave's 16 rows (ISA 7.12.2 16-bit A layout)
        const unsigned short* ar = &lA[buf][(wave * 16 + ln15) * 32];
        us8 a0 = *(const us8*)(ar + hi * 8);
        us8 a1 = *(const us8*)(ar + 16 + hi * 8);
        us8 b0[4], b1[4];
#pragma unroll
        for (int j = 0; j < 4; ++j) {
            const unsigned short* br = &lB[buf][(j * 16 + ln15) * 32];
            b0[j] = *(const us8*)(br + hi * 8);
            b1[j] = *(const us8*)(br + 16 + hi * 8);
        }

        v16bf av;
#pragma unroll
        for (int i = 0; i < 8; ++i) { av[i] = us2bf(a0.v[i]); av[8 + i] = us2bf(a1.v[i]); }

        // ---- 4 independent WMMAs back-to-back (no D->A/B hazard chain) ----
#pragma unroll
        for (int j = 0; j < 4; ++j) {
            v16bf bv;
#pragma unroll
            for (int i = 0; i < 8; ++i) { bv[i] = us2bf(b0[j].v[i]); bv[8 + i] = us2bf(b1[j].v[i]); }
            acc[j] = __builtin_amdgcn_wmma_f32_16x16x32_bf16(
                false, av, false, bv, (short)0, acc[j], false, false);
        }
        buf ^= 1;
    }

#pragma unroll
    for (int j = 0; j < 4; ++j) {
#pragma unroll
        for (int r = 0; r < 8; ++r) {
            int row = tileM + wave * 16 + hi * 8 + r;
            int col = tileN + j * 16 + ln15;
            if (row >= M || col >= N) continue;
            size_t idx = (size_t)z * strC + (size_t)row * N + col;
            float val = acc[j][r];
            if (EPI == 0) {
                if (bias)   val += bias[col];
                if (addend) val += addend[idx];
                Cf[idx] = val;
            } else if (EPI == 1) {
                val += bias[col];
                float g = 0.5f * val * (1.0f + tanhf(0.7978845608028654f *
                                                     (val + 0.044715f * val * val * val)));
                Cb[idx] = f2bf(g);
            } else if (EPI == 2) {
                bool x0q = row >= Lc, x0k = col >= Lc;
                int bq = (x0q ? row - Lc : row) >> 2;
                int bk = (x0k ? col - Lc : col) >> 2;
                bool can = (bq == bk && x0q == x0k) ||
                           (bq >  bk && x0k && !x0q) ||
                           (bq >= bk && x0k &&  x0q);
                if (!can) val = -1e30f;
                Cb[idx] = f2bf(val);
            } else { // EPI == 3
                int b = z / Hc;
                if (emask[b * Sc + col] == 0) val = -1e30f;
                Cb[idx] = f2bf(val);
            }
        }
    }
}

// ---------------- host orchestration ----------------
extern "C" void kernel_launch(void* const* d_in, const int* in_sizes, int n_in,
                              void* d_out, int out_size, void* d_ws, size_t ws_size,
                              hipStream_t stream) {
    (void)in_sizes; (void)n_in; (void)out_size; (void)ws_size;

    const int*   ids      = (const int*)  d_in[0];
    const float* enc      = (const float*)d_in[1];
    const int*   emask    = (const int*)  d_in[2];
    const float* emb      = (const float*)d_in[3];
    const float* pos      = (const float*)d_in[4];
    const float* ln_emb_s = (const float*)d_in[5];
    const float* ln_emb_b = (const float*)d_in[6];
    const float* attn_w   = (const float*)d_in[7];
    const float* attn_b   = (const float*)d_in[8];
    const float* ln_w     = (const float*)d_in[9];
    const float* ln_b     = (const float*)d_in[10];
    const float* fc1_w    = (const float*)d_in[11];
    const float* fc1_b    = (const float*)d_in[12];
    const float* fc2_w    = (const float*)d_in[13];
    const float* fc2_b    = (const float*)d_in[14];
    const float* fln_s    = (const float*)d_in[15];
    const float* fln_b    = (const float*)d_in[16];
    const float* lm_w     = (const float*)d_in[17];
    float* out = (float*)d_out;

    char* ws = (char*)d_ws;
    size_t off = 0;
    auto alloc = [&](size_t bytes) -> void* {
        void* p = ws + off;
        off = (off + bytes + 255) & ~(size_t)255;
        return p;
    };

    const size_t nAttnW = (size_t)NLc * 2 * 4 * Dc * Dc;
    const size_t nFc1   = (size_t)NLc * DFFc * Dc;
    const size_t nFc2   = (size_t)NLc * Dc * DFFc;
    const size_t nLm    = (size_t)Vc * Dc;
    const size_t nEnc   = (size_t)Bc * Sc * Dc;
    const size_t nBTD   = (size_t)Bc * Tc * Dc;

    unsigned short* attn_bf = (unsigned short*)alloc(nAttnW * 2);
    unsigned short* fc1_bf  = (unsigned short*)alloc(nFc1 * 2);
    unsigned short* fc2_bf  = (unsigned short*)alloc(nFc2 * 2);
    unsigned short* lm_bf   = (unsigned short*)alloc(nLm * 2);
    unsigned short* enc_bf  = (unsigned short*)alloc(nEnc * 2);
    float*          h       = (float*)         alloc(nBTD * 4);
    unsigned short* xbf     = (unsigned short*)alloc(nBTD * 2);
    float*          qf      = (float*)         alloc(nBTD * 4);
    float*          kf      = (float*)         alloc(nBTD * 4);
    float*          vf      = (float*)         alloc(nBTD * 4);
    unsigned short* qh      = (unsigned short*)alloc(nBTD * 2);
    unsigned short* kh      = (unsigned short*)alloc(nBTD * 2);
    unsigned short* vhT     = (unsigned short*)alloc(nBTD * 2);
    unsigned short* probs   = (unsigned short*)alloc((size_t)Bc * Hc * Tc * Tc * 2);
    float*          ctx     = (float*)         alloc(nBTD * 4);
    unsigned short* ctxbf   = (unsigned short*)alloc(nBTD * 2);
    unsigned short* midbf   = (unsigned short*)alloc((size_t)Bc * Tc * DFFc * 2);

    auto cdiv = [](long long a, long long b) { return (unsigned)((a + b - 1) / b); };
    auto cvt = [&](const float* s, unsigned short* d, size_t n) {
        cvt_f32_bf16_kernel<<<cdiv((long long)n, 256), 256, 0, stream>>>(s, d, n);
    };

    auto gemm_f32 = [&](const unsigned short* A, const unsigned short* Bw,
                        const float* bias, const float* addend, float* Cf,
                        int M, int N, int K, int batch,
                        long long sA, long long sB, long long sC) {
        dim3 g(cdiv(N, 64), cdiv(M, 128), batch);
        wmma_gemm_kernel<0><<<g, 256, 0, stream>>>(A, Bw, bias, addend, Cf, nullptr,
                                                   nullptr, M, N, K, sA, sB, sC);
    };
    auto gemm_gelu = [&](const unsigned short* A, const unsigned short* Bw,
                         const float* bias, unsigned short* Cb, int M, int N, int K) {
        dim3 g(cdiv(N, 64), cdiv(M, 128), 1);
        wmma_gemm_kernel<1><<<g, 256, 0, stream>>>(A, Bw, bias, nullptr, nullptr, Cb,
                                                   nullptr, M, N, K, 0, 0, 0);
    };
    auto gemm_selfmask = [&](const unsigned short* A, const unsigned short* Bw,
                             unsigned short* Cb, int M, int N, int K, int batch,
                             long long sA, long long sB, long long sC) {
        dim3 g(cdiv(N, 64), cdiv(M, 128), batch);
        wmma_gemm_kernel<2><<<g, 256, 0, stream>>>(A, Bw, nullptr, nullptr, nullptr, Cb,
                                                   nullptr, M, N, K, sA, sB, sC);
    };
    auto gemm_crossmask = [&](const unsigned short* A, const unsigned short* Bw,
                              unsigned short* Cb, int M, int N, int K, int batch,
                              long long sA, long long sB, long long sC) {
        dim3 g(cdiv(N, 64), cdiv(M, 128), batch);
        wmma_gemm_kernel<3><<<g, 256, 0, stream>>>(A, Bw, nullptr, nullptr, nullptr, Cb,
                                                   emask, M, N, K, sA, sB, sC);
    };

    // ---- weight / encoder down-conversion (deterministic, every launch) ----
    cvt(attn_w, attn_bf, nAttnW);
    cvt(fc1_w,  fc1_bf,  nFc1);
    cvt(fc2_w,  fc2_bf,  nFc2);
    cvt(lm_w,   lm_bf,   nLm);
    cvt(enc,    enc_bf,  nEnc);

    // ---- embedding + LN -> residual stream h (fp32) ----
    embed_ln_kernel<<<Bc * Tc, 256, 0, stream>>>(ids, emb, pos, ln_emb_s, ln_emb_b, h);

    const int MT  = Bc * Tc;   // 4096 rows
    const int MS  = Bc * Sc;   // 512 rows
    const int BH  = Bc * Hc;   // 64 head-batches
    const unsigned split_grid_T = cdiv((long long)Bc * Tc * Dc, 256);
    const unsigned split_grid_S = cdiv((long long)Bc * Sc * Dc, 256);

    for (int l = 0; l < NLc; ++l) {
        const unsigned short* Wself = attn_bf + (((size_t)l * 2 + 0) * 4) * Dc * Dc;
        const unsigned short* Wcross= attn_bf + (((size_t)l * 2 + 1) * 4) * Dc * Dc;
        const float* bself  = attn_b + (((size_t)l * 2 + 0) * 4) * Dc;
        const float* bcross = attn_b + (((size_t)l * 2 + 1) * 4) * Dc;

        // ================= self attention =================
        layernorm_bf16_kernel<<<MT, 256, 0, stream>>>(h, ln_w + ((size_t)l*3+0)*Dc,
                                                      ln_b + ((size_t)l*3+0)*Dc, xbf);
        gemm_f32(xbf, Wself + 0*(size_t)Dc*Dc, bself + 0*Dc, nullptr, qf, MT, Dc, Dc, 1, 0,0,0);
        gemm_f32(xbf, Wself + 1*(size_t)Dc*Dc, bself + 1*Dc, nullptr, kf, MT, Dc, Dc, 1, 0,0,0);
        gemm_f32(xbf, Wself + 2*(size_t)Dc*Dc, bself + 2*Dc, nullptr, vf, MT, Dc, Dc, 1, 0,0,0);

        split_heads_kernel  <<<split_grid_T, 256, 0, stream>>>(qf, qh, Tc, 0.125f);
        split_heads_kernel  <<<split_grid_T, 256, 0, stream>>>(kf, kh, Tc, 1.0f);
        split_heads_T_kernel<<<split_grid_T, 256, 0, stream>>>(vf, vhT, Tc);

        gemm_selfmask(qh, kh, probs, Tc, Tc, HDc, BH,
                      (long long)Tc*HDc, (long long)Tc*HDc, (long long)Tc*Tc);
        softmax_bf16_kernel<<<BH * Tc, 256, 0, stream>>>(probs, Tc);
        gemm_f32(probs, vhT, nullptr, nullptr, ctx, Tc, HDc, Tc, BH,
                 (long long)Tc*Tc, (long long)HDc*Tc, (long long)Tc*HDc);
        merge_heads_kernel<<<split_grid_T, 256, 0, stream>>>(ctx, ctxbf);
        gemm_f32(ctxbf, Wself + 3*(size_t)Dc*Dc, bself + 3*Dc, h, h, MT, Dc, Dc, 1, 0,0,0);

        // ================= cross attention =================
        layernorm_bf16_kernel<<<MT, 256, 0, stream>>>(h, ln_w + ((size_t)l*3+1)*Dc,
                                                      ln_b + ((size_t)l*3+1)*Dc, xbf);
        gemm_f32(xbf,    Wcross + 0*(size_t)Dc*Dc, bcross + 0*Dc, nullptr, qf, MT, Dc, Dc, 1, 0,0,0);
        gemm_f32(enc_bf, Wcross + 1*(size_t)Dc*Dc, bcross + 1*Dc, nullptr, kf, MS, Dc, Dc, 1, 0,0,0);
        gemm_f32(enc_bf, Wcross + 2*(size_t)Dc*Dc, bcross + 2*Dc, nullptr, vf, MS, Dc, Dc, 1, 0,0,0);

        split_heads_kernel  <<<split_grid_T, 256, 0, stream>>>(qf, qh, Tc, 0.125f);
        split_heads_kernel  <<<split_grid_S, 256, 0, stream>>>(kf, kh, Sc, 1.0f);
        split_heads_T_kernel<<<split_grid_S, 256, 0, stream>>>(vf, vhT, Sc);

        gemm_crossmask(qh, kh, probs, Tc, Sc, HDc, BH,
                       (long long)Tc*HDc, (long long)Sc*HDc, (long long)Tc*Sc);
        softmax_bf16_kernel<<<BH * Tc, 256, 0, stream>>>(probs, Sc);
        gemm_f32(probs, vhT, nullptr, nullptr, ctx, Tc, HDc, Sc, BH,
                 (long long)Tc*Sc, (long long)HDc*Sc, (long long)Tc*HDc);
        merge_heads_kernel<<<split_grid_T, 256, 0, stream>>>(ctx, ctxbf);
        gemm_f32(ctxbf, Wcross + 3*(size_t)Dc*Dc, bcross + 3*Dc, h, h, MT, Dc, Dc, 1, 0,0,0);

        // ================= FFN =================
        layernorm_bf16_kernel<<<MT, 256, 0, stream>>>(h, ln_w + ((size_t)l*3+2)*Dc,
                                                      ln_b + ((size_t)l*3+2)*Dc, xbf);
        gemm_gelu(xbf, fc1_bf + (size_t)l*DFFc*Dc, fc1_b + (size_t)l*DFFc, midbf, MT, DFFc, Dc);
        gemm_f32(midbf, fc2_bf + (size_t)l*Dc*DFFc, fc2_b + (size_t)l*Dc, h, h, MT, Dc, DFFc, 1, 0,0,0);
    }

    // ---- final LN + LM head ----
    layernorm_bf16_kernel<<<MT, 256, 0, stream>>>(h, fln_s, fln_b, xbf);
    gemm_f32(xbf, lm_bf, nullptr, nullptr, out, MT, Vc, Dc, 1, 0, 0, 0);
}
